// MultiHeadAttention_38646115729512
// MI455X (gfx1250) — matrix operand order repaired and checked by reading, not compiled, not run on hardware
//
#include <hip/hip_runtime.h>
#include <math.h>

// Problem constants (match reference)
#define BB   4
#define SS   1024
#define DD   1024
#define HH   16
#define DK   64

typedef __attribute__((ext_vector_type(16))) __bf16 v16bf;
typedef __attribute__((ext_vector_type(8)))  __bf16 v8bf;
typedef __attribute__((ext_vector_type(8)))  float  v8f;
typedef unsigned int u32x4 __attribute__((ext_vector_type(4)));
typedef int          i32x4 __attribute__((ext_vector_type(4)));
typedef int          i32x8 __attribute__((ext_vector_type(8)));

#if __has_builtin(__builtin_amdgcn_tensor_load_to_lds) && \
    __has_builtin(__builtin_amdgcn_s_wait_tensorcnt)
#define HAVE_TDM 1
#else
#define HAVE_TDM 0
#endif

union FragA { v16bf v; v8bf h[2]; };

// fp32 -> bf16, round-to-nearest-even
__device__ __forceinline__ unsigned short f2bf(float f) {
    union { float f; unsigned u; } c; c.f = f;
    unsigned r = c.u + 0x7FFFu + ((c.u >> 16) & 1u);
    return (unsigned short)(r >> 16);
}

#if HAVE_TDM
// Issue one TDM 2D tile load (16-bit elements) into LDS.
// D# layout per CDNA5 ISA 8.3-8.6: group0 = {count/lds_addr/global_addr/type},
// group1 = {data_size, tensor dims, tile dims, dim0 stride}. Tensor dims are
// set huge (tiles are always fully in-bounds here).
__device__ __forceinline__ void tdm_load_tile_b16(
    unsigned lds_off, const void* gsrc, int tile_cols, int tile_rows,
    unsigned long long row_stride_elems)
{
    unsigned long long ga = (unsigned long long)gsrc;
    u32x4 g0;
    g0[0] = 1u;                                   // count=1 (valid), user mode
    g0[1] = lds_off;                              // lds_addr (bytes)
    g0[2] = (unsigned)ga;                         // global_addr[31:0]
    g0[3] = ((unsigned)(ga >> 32) & 0x01FFFFFFu)  // global_addr[56:32]
            | 0x80000000u;                        // type=2 ("image")
    const unsigned td = 0x40000000u;              // huge tensor dims (no OOB)
    i32x8 g1;
    g1[0] = (int)(1u << 16);                      // data_size=1 -> 2 bytes
    g1[1] = (int)((td & 0xFFFFu) << 16);                          // tensor_dim0 lo
    g1[2] = (int)(((td >> 16) & 0xFFFFu) | ((td & 0xFFFFu) << 16)); // dim0 hi | dim1 lo
    g1[3] = (int)(((td >> 16) & 0xFFFFu) | ((unsigned)tile_cols << 16)); // dim1 hi | tile_dim0
    g1[4] = (int)(tile_rows & 0xFFFF);            // tile_dim1 (tile_dim2=0)
    g1[5] = (int)(row_stride_elems & 0xFFFFFFFFu);        // dim0_stride lo
    g1[6] = (int)((row_stride_elems >> 32) & 0xFFFFu);    // dim0_stride hi
    g1[7] = 0;
    i32x4 g2; g2[0] = (int)td; g2[1] = (int)td; g2[2] = 0; g2[3] = 0;
    i32x4 g3; g3[0] = 0; g3[1] = (int)td; g3[2] = 0; g3[3] = 0;
#if defined(__clang_major__) && (__clang_major__ >= 23)
    i32x8 g4 = {};
    __builtin_amdgcn_tensor_load_to_lds(g0, g1, g2, g3, g4, 0);
#else
    __builtin_amdgcn_tensor_load_to_lds(g0, g1, g2, g3, 0);
#endif
}
#endif

// ---------------------------------------------------------------------------
// Unified GEMM: C = scale * A(MxK) x B(NxK)^T, bf16 WMMA, fp32 accumulate.
//   B is ALWAYS bf16 row-major [N,K]  (weights pre-transposed; k rows; v^T).
//   A is bf16 [M,K] (ABF16=true) or fp32 [M,K] (false, VALU-converted).
//   OUTMODE: 0 = fp32, 1 = bf16, 2 = bf16 written v-transposed [b][n][s].
// Macro tile 128x64, BK=32, 256 thr = 8 waves, wave tile 32x32 (2x2 WMMA).
// Double-buffered LDS; bf16 tiles staged by the Tensor Data Mover (wave 0
// issues, TENSORcnt-waited, 1 barrier/iter) with a manual-copy fallback.
// ---------------------------------------------------------------------------
template <bool ABF16, int OUTMODE>
__global__ __launch_bounds__(256) void gemm_wmma(
    int M, int N, int K,
    const void* __restrict__ Ap, int lda, long sAb, long sAh,
    const unsigned short* __restrict__ Bp, int ldb, long sBb, long sBh,
    void* __restrict__ Cp, int ldc, long sCb, long sCh,
    int Hdiv, float scale)
{
    __shared__ __attribute__((aligned(32))) unsigned short As[2][128 * 32];
    __shared__ __attribute__((aligned(32))) unsigned short Bts[2][64 * 32];

    const int z = blockIdx.z, zb = z / Hdiv, zh = z % Hdiv;
    const unsigned short* Bb = Bp + (long)zb * sBb + (long)zh * sBh;
    const unsigned short* Ab16 =
        ABF16 ? (const unsigned short*)Ap + (long)zb * sAb + (long)zh * sAh : nullptr;
    const float* Ab32 =
        ABF16 ? nullptr : (const float*)Ap + (long)zb * sAb + (long)zh * sAh;

    const int m0 = blockIdx.y * 128, n0 = blockIdx.x * 64;
    const int tid = threadIdx.x, lane = tid & 31, wave = tid >> 5;
    const int wm = wave & 3, wn = wave >> 2;
    const bool w0 = (wave == 0);
    (void)w0;

#if HAVE_TDM
    const unsigned asOff[2] = {
        (unsigned)(unsigned long long)(const void*)&As[0][0],
        (unsigned)(unsigned long long)(const void*)&As[1][0] };
    const unsigned btOff[2] = {
        (unsigned)(unsigned long long)(const void*)&Bts[0][0],
        (unsigned)(unsigned long long)(const void*)&Bts[1][0] };
#endif

    // Early part of staging (TDM issues; overlap DMA with WMMA below)
    auto stage_tdm = [&](int buf, int k0) {
#if HAVE_TDM
        if (w0) {
            tdm_load_tile_b16(btOff[buf], Bb + (long)n0 * ldb + k0, 32, 64,
                              (unsigned long long)ldb);
            if (ABF16)
                tdm_load_tile_b16(asOff[buf], Ab16 + (long)m0 * lda + k0, 32, 128,
                                  (unsigned long long)lda);
        }
#else
        (void)buf; (void)k0;
#endif
    };
    // Late part of staging (manual copies; issued after WMMAs of current step)
    auto stage_manual = [&](int buf, int k0) {
#if !HAVE_TDM
        {   // B tile: 64x32 bf16 = 256 x 16B chunks
            int n = tid >> 2, c8 = (tid & 3) << 3;
            *(uint4*)(&Bts[buf][n * 32 + c8]) =
                *(const uint4*)(Bb + (long)(n0 + n) * ldb + k0 + c8);
        }
        if (ABF16) {  // A tile: 128x32 bf16 = 512 x 16B chunks
            #pragma unroll
            for (int i = 0; i < 2; ++i) {
                int chunk = tid + 256 * i;
                int r = chunk >> 2, c8 = (chunk & 3) << 3;
                *(uint4*)(&As[buf][r * 32 + c8]) =
                    *(const uint4*)(Ab16 + (long)(m0 + r) * lda + k0 + c8);
            }
        }
#endif
        if (!ABF16) { // A tile from fp32: load, convert, store
            #pragma unroll
            for (int i = 0; i < 4; ++i) {
                int chunk = tid + 256 * i;
                int r = chunk >> 3, c4 = (chunk & 7) << 2;
                float4 f = *(const float4*)(Ab32 + (long)(m0 + r) * lda + k0 + c4);
                ushort4 u;
                u.x = f2bf(f.x); u.y = f2bf(f.y); u.z = f2bf(f.z); u.w = f2bf(f.w);
                *(ushort4*)(&As[buf][r * 32 + c4]) = u;
            }
        }
        (void)buf; (void)k0;
    };

    v8f acc[2][2] = {};
    const int nk = K / 32;

    stage_tdm(0, 0);
    stage_manual(0, 0);

    for (int kt = 0; kt < nk; ++kt) {
        const int buf = kt & 1;
#if HAVE_TDM
        if (w0) __builtin_amdgcn_s_wait_tensorcnt(0);
#endif
        __syncthreads();

        const bool more = (kt + 1) < nk;
        if (more) stage_tdm(buf ^ 1, (kt + 1) * 32);

        const unsigned short* A_ = As[buf];
        const unsigned short* B_ = Bts[buf];
        FragA a0, a1;
        {   // A 16x32 layout: lane half -> K 0-7/8-15; upper v8 holds K+16
            int m = wm * 32 + (lane & 15);
            int kb2 = (lane >> 4) << 3;
            a0.h[0] = *(const v8bf*)(A_ + m * 32 + kb2);
            a0.h[1] = *(const v8bf*)(A_ + m * 32 + kb2 + 16);
            a1.h[0] = *(const v8bf*)(A_ + (m + 16) * 32 + kb2);
            a1.h[1] = *(const v8bf*)(A_ + (m + 16) * 32 + kb2 + 16);
        }
        FragA b0, b1;
        {   // B layout: same as A (lane half -> K 0-7/8-15; upper v8 holds K+16)
            int n = wn * 32 + (lane & 15);
            int kb2 = (lane >> 4) << 3;
            b0.h[0] = *(const v8bf*)(B_ + n * 32 + kb2);
            b0.h[1] = *(const v8bf*)(B_ + n * 32 + kb2 + 16);
            b1.h[0] = *(const v8bf*)(B_ + (n + 16) * 32 + kb2);
            b1.h[1] = *(const v8bf*)(B_ + (n + 16) * 32 + kb2 + 16);
        }
        acc[0][0] = __builtin_amdgcn_wmma_f32_16x16x32_bf16(false, a0.v, false, b0.v, (short)0, acc[0][0], false, false);
        acc[0][1] = __builtin_amdgcn_wmma_f32_16x16x32_bf16(false, a0.v, false, b1.v, (short)0, acc[0][1], false, false);
        acc[1][0] = __builtin_amdgcn_wmma_f32_16x16x32_bf16(false, a1.v, false, b0.v, (short)0, acc[1][0], false, false);
        acc[1][1] = __builtin_amdgcn_wmma_f32_16x16x32_bf16(false, a1.v, false, b1.v, (short)0, acc[1][1], false, false);

        if (more) stage_manual(buf ^ 1, (kt + 1) * 32);
    }

    // Epilogue (C/D layout: VGPR j -> M = j + 8*(lane>=16), N = lane&15)
    float* Cf = (float*)Cp + (long)zb * sCb + (long)zh * sCh;
    unsigned short* Ch = (unsigned short*)Cp + (long)zb * sCb + (long)zh * sCh;
    #pragma unroll
    for (int mi = 0; mi < 2; ++mi)
        #pragma unroll
        for (int ni = 0; ni < 2; ++ni) {
            int n  = n0 + wn * 32 + ni * 16 + (lane & 15);
            int mb = m0 + wm * 32 + mi * 16 + ((lane >> 4) << 3);
            #pragma unroll
            for (int j = 0; j < 8; ++j) {
                float r = scale * acc[mi][ni][j];
                if (OUTMODE == 0) {
                    Cf[(long)(mb + j) * ldc + n] = r;
                } else if (OUTMODE == 1) {
                    Ch[(long)(mb + j) * ldc + n] = f2bf(r);
                } else { // v-transposed: [b][n][s], m = b*1024 + s
                    int mr = mb + j;
                    long off = (long)(mr >> 10) * (1024 * 1024) + (long)n * 1024 + (mr & 1023);
                    ((unsigned short*)Cp)[off] = f2bf(r);
                }
            }
        }
}

// ---------------------------------------------------------------------------
// fp32 -> bf16 bulk convert (4 elements / thread, exact-size launch)
// ---------------------------------------------------------------------------
__global__ __launch_bounds__(256) void cvt_f32_to_bf16(
    const float* __restrict__ in, unsigned short* __restrict__ out)
{
    long i = ((long)blockIdx.x * 256 + threadIdx.x) * 4;
    float4 f = *(const float4*)(in + i);
    ushort4 u;
    u.x = f2bf(f.x); u.y = f2bf(f.y); u.z = f2bf(f.z); u.w = f2bf(f.w);
    *(ushort4*)(out + i) = u;
}

// ---------------------------------------------------------------------------
// 1024x1024 transpose + convert: W[K,N] fp32 -> WT[N,K] bf16. 32x32 LDS tiles.
// ---------------------------------------------------------------------------
__global__ __launch_bounds__(256) void transpose_cvt_1024(
    const float* __restrict__ W, unsigned short* __restrict__ WT)
{
    __shared__ float t[32][33];
    int bx = blockIdx.x * 32;  // column tile of W (n)
    int by = blockIdx.y * 32;  // row tile of W (k)
    int x = threadIdx.x, y = threadIdx.y;
    #pragma unroll
    for (int i = 0; i < 32; i += 8)
        t[y + i][x] = W[(long)(by + y + i) * 1024 + bx + x];
    __syncthreads();
    #pragma unroll
    for (int i = 0; i < 32; i += 8)
        WT[(long)(bx + y + i) * 1024 + by + x] = f2bf(t[x][y + i]);
}

// ---------------------------------------------------------------------------
// Softmax over the QUERY axis (axis=2): for fixed (b,h,ki) normalize over qi.
// In place on the fp32 attn output buffer.
// ---------------------------------------------------------------------------
__global__ __launch_bounds__(256) void softmax_qaxis(
    float* __restrict__ attn, const unsigned char* __restrict__ mask)
{
    const int bid    = blockIdx.x;
    const int bh     = bid >> 6;       // / (S/16)
    const int colblk = bid & 63;
    const int b      = bh >> 4;        // / H
    const int t = threadIdx.x;
    const int c = t & 15;
    const int g = t >> 4;
    const int ki = colblk * 16 + c;

    float* base = attn + (long)bh * SS * SS;
    const unsigned char* mb = mask + (long)b * SS * SS;

    float vals[64];
    float mloc = -INFINITY;
    #pragma unroll 4
    for (int j = 0; j < 64; ++j) {
        int qi = g + 16 * j;
        long idx = (long)qi * SS + ki;
        float v = base[idx];
        if (mb[idx]) v = -INFINITY;
        vals[j] = v;
        mloc = fmaxf(mloc, v);
    }

    __shared__ float red[16][17];
    red[g][c] = mloc;
    __syncthreads();
    for (int off = 8; off > 0; off >>= 1) {
        if (g < off) red[g][c] = fmaxf(red[g][c], red[g + off][c]);
        __syncthreads();
    }
    float cmax = red[0][c];
    __syncthreads();

    float sloc = 0.f;
    #pragma unroll 4
    for (int j = 0; j < 64; ++j) {
        vals[j] = __expf(vals[j] - cmax);
        sloc += vals[j];
    }
    red[g][c] = sloc;
    __syncthreads();
    for (int off = 8; off > 0; off >>= 1) {
        if (g < off) red[g][c] += red[g + off][c];
        __syncthreads();
    }
    float inv = 1.0f / red[0][c];

    #pragma unroll 4
    for (int j = 0; j < 64; ++j) {
        int qi = g + 16 * j;
        base[(long)qi * SS + ki] = vals[j] * inv;
    }
}

// ---------------------------------------------------------------------------
// Fused residual add + LayerNorm over last dim (1024). Block per row.
// ---------------------------------------------------------------------------
__global__ __launch_bounds__(256) void resid_layernorm(
    const float* __restrict__ x, const float* __restrict__ resid,
    const float* __restrict__ gamma, const float* __restrict__ beta,
    float* __restrict__ out)
{
    const long row = blockIdx.x;
    const int t = threadIdx.x;
    float v[4];
    float s = 0.f, ss = 0.f;
    #pragma unroll
    for (int i = 0; i < 4; ++i) {
        int c = t + 256 * i;
        float a = x[row * DD + c] + resid[row * DD + c];
        v[i] = a;
        s += a;
        ss += a * a;
    }
    __shared__ float rs[256], rss[256];
    rs[t] = s; rss[t] = ss;
    __syncthreads();
    for (int off = 128; off > 0; off >>= 1) {
        if (t < off) { rs[t] += rs[t + off]; rss[t] += rss[t + off]; }
        __syncthreads();
    }
    float mean = rs[0] * (1.0f / DD);
    float var  = rss[0] * (1.0f / DD) - mean * mean;
    float inv  = rsqrtf(var + 1e-5f);
    #pragma unroll
    for (int i = 0; i < 4; ++i) {
        int c = t + 256 * i;
        out[row * DD + c] = (v[i] - mean) * inv * gamma[c] + beta[c];
    }
}

// ---------------------------------------------------------------------------
extern "C" void kernel_launch(void* const* d_in, const int* in_sizes, int n_in,
                              void* d_out, int out_size, void* d_ws, size_t ws_size,
                              hipStream_t stream)
{
    (void)in_sizes; (void)n_in; (void)out_size; (void)ws_size;

    const float* Q    = (const float*)d_in[0];
    const float* Kin  = (const float*)d_in[1];
    const float* Vin  = (const float*)d_in[2];
    const unsigned char* mask = (const unsigned char*)d_in[3];
    const float* W_Q  = (const float*)d_in[4];
    const float* W_K  = (const float*)d_in[5];
    const float* W_V  = (const float*)d_in[6];
    const float* W_O  = (const float*)d_in[7];
    const float* gam  = (const float*)d_in[8];
    const float* bet  = (const float*)d_in[9];

    const long P  = (long)BB * SS * DD;   // 4,194,304 activation elems
    const long W1 = (long)DD * DD;        // 1,048,576 weight elems
    float* out  = (float*)d_out;
    float* attn = out + P;                // attn output region (B,H,S,S) fp32

    unsigned short* u = (unsigned short*)d_ws;
    unsigned short* Qb  = u;              // bf16 inputs
    unsigned short* Kb  = u + P;
    unsigned short* Vb  = u + 2 * P;
    unsigned short* qb  = u + 3 * P;      // q projection (bf16)
    unsigned short* kb  = u + 4 * P;      // k projection (bf16)
    unsigned short* vbT = u + 5 * P;      // v projection, transposed [b][hd][s]
    unsigned short* ao  = u + 6 * P;      // attn @ v (bf16)
    unsigned short* wqT = u + 7 * P;      // weights, transposed bf16 [N,K]
    unsigned short* wkT = wqT + W1;
    unsigned short* wvT = wqT + 2 * W1;
    unsigned short* woT = wqT + 3 * W1;
    float* po = (float*)(u + 8 * P);      // (attn@v) @ W_O  fp32

    dim3 blk(256);
    dim3 tb(32, 8);

    // ---- precision prep: bf16 activations + transposed bf16 weights
    cvt_f32_to_bf16<<<dim3(P / 1024), blk, 0, stream>>>(Q,   Qb);
    cvt_f32_to_bf16<<<dim3(P / 1024), blk, 0, stream>>>(Kin, Kb);
    cvt_f32_to_bf16<<<dim3(P / 1024), blk, 0, stream>>>(Vin, Vb);
    transpose_cvt_1024<<<dim3(32, 32), tb, 0, stream>>>(W_Q, wqT);
    transpose_cvt_1024<<<dim3(32, 32), tb, 0, stream>>>(W_K, wkT);
    transpose_cvt_1024<<<dim3(32, 32), tb, 0, stream>>>(W_V, wvT);
    transpose_cvt_1024<<<dim3(32, 32), tb, 0, stream>>>(W_O, woT);

    // ---- input projections (bf16 out; v stored transposed per head-dim)
    gemm_wmma<true, 1><<<dim3(16, 32, 1), blk, 0, stream>>>(
        4096, 1024, 1024, Qb, 1024, 0, 0, wqT, 1024, 0, 0, qb, 1024, 0, 0, 1, 1.0f);
    gemm_wmma<true, 1><<<dim3(16, 32, 1), blk, 0, stream>>>(
        4096, 1024, 1024, Kb, 1024, 0, 0, wkT, 1024, 0, 0, kb, 1024, 0, 0, 1, 1.0f);
    gemm_wmma<true, 2><<<dim3(16, 32, 1), blk, 0, stream>>>(
        4096, 1024, 1024, Vb, 1024, 0, 0, wvT, 1024, 0, 0, vbT, 1024, 0, 0, 1, 1.0f);

    // ---- scores: per (b,h)  q x k^T -> fp32 attn, scaled 1/sqrt(64)
    gemm_wmma<true, 0><<<dim3(16, 8, BB * HH), blk, 0, stream>>>(
        1024, 1024, 64,
        qb, 1024, (long)SS * DD, DK,
        kb, 1024, (long)SS * DD, DK,
        attn, 1024, (long)HH * SS * SS, (long)SS * SS,
        HH, 0.125f);

    // ---- query-axis softmax, in place, with mask
    softmax_qaxis<<<dim3(BB * HH * (SS / 16)), blk, 0, stream>>>(attn, mask);

    // ---- attn @ v: A = attn fp32, B = v^T bf16 [64,1024] per (b,h)
    gemm_wmma<false, 1><<<dim3(1, 8, BB * HH), blk, 0, stream>>>(
        1024, 64, 1024,
        attn, 1024, (long)HH * SS * SS, (long)SS * SS,
        vbT, 1024, (long)SS * DD, (long)DK * SS,
        ao, 1024, (long)SS * DD, DK,
        HH, 1.0f);

    // ---- output projection -> fp32
    gemm_wmma<true, 0><<<dim3(16, 32, 1), blk, 0, stream>>>(
        4096, 1024, 1024, ao, 1024, 0, 0, woT, 1024, 0, 0, po, 1024, 0, 0, 1, 1.0f);

    // ---- residual + LayerNorm -> out
    resid_layernorm<<<dim3(BB * SS), blk, 0, stream>>>(po, Q, gam, bet, out);
}
